// OptimizedMiniBlock_21861383537484
// MI455X (gfx1250) — matrix operand order repaired and checked by reading, not compiled, not run on hardware
//
#include <hip/hip_runtime.h>

// ---------------------------------------------------------------------------
// Problem constants (from reference setup_inputs): B=4, S=2048, D=1024
// ---------------------------------------------------------------------------
constexpr int Bc = 4;
constexpr int Sc = 2048;
constexpr int Dc = 1024;

typedef __attribute__((ext_vector_type(16))) __bf16 v16bf;
typedef __attribute__((ext_vector_type(8)))  float  v8f;
typedef __attribute__((ext_vector_type(4)))  unsigned int v4u;
typedef __attribute__((ext_vector_type(8)))  unsigned int v8u;

typedef __attribute__((address_space(3))) char* lds_raw_t;

__device__ __forceinline__ v8f zero8() {
    v8f z;
#pragma unroll
    for (int i = 0; i < 8; ++i) z[i] = 0.0f;
    return z;
}

__device__ __forceinline__ v8f wmma_bf16(v16bf a, v16bf b, v8f c) {
    // D = A(16x32 bf16) * B(32x16 bf16) + C(16x16 f32)
    return __builtin_amdgcn_wmma_f32_16x16x32_bf16(
        /*neg_a=*/false, a, /*neg_b=*/false, b,
        /*c_mod=*/(short)0, c, /*reuse_a=*/false, /*reuse_b=*/false);
}

// Async copy of 16 bytes global -> LDS (CDNA5 GLOBAL_LOAD_ASYNC_TO_LDS_B128).
__device__ __forceinline__ void async_copy_b128(void* lds_dst, const void* gsrc) {
    unsigned int lds_off =
        (unsigned int)(unsigned long long)(lds_raw_t)lds_dst;   // AS3 offset
    unsigned long long ga = (unsigned long long)gsrc;
    asm volatile("global_load_async_to_lds_b128 %0, %1, off"
                 :: "v"(lds_off), "v"(ga)
                 : "memory");
}

__device__ __forceinline__ void wait_async_zero() {
    asm volatile("s_wait_asynccnt 0x0" ::: "memory");
}

// ---------------------------------------------------------------------------
// Tensor Data Mover: 1-D contiguous copy of `elems` 2-byte elements from
// global memory to LDS. Descriptor per CDNA5 ISA §8.3/§8.4:
//   group0: count=1 | lds_addr | global_addr[56:0] | type=2
//   group1: data_size=2B, tensor_dim0=tile_dim0=elems (elems<=65535),
//           tensor_dim1=tile_dim1=1, dim0 stride=elems, no multicast/pad.
// 2-group form (VADDR2/VADDR3 omitted = NULL) is valid for <=2-D tensors.
// ---------------------------------------------------------------------------
__device__ __forceinline__ void tdm_load_to_lds_1d(unsigned int lds_off,
                                                   unsigned long long gaddr,
                                                   unsigned int elems) {
    v4u g0;
    g0[0] = 1u;                                               // count=1
    g0[1] = lds_off;                                          // lds_addr
    g0[2] = (unsigned int)(gaddr & 0xFFFFFFFFull);            // gaddr lo
    g0[3] = (unsigned int)((gaddr >> 32) & 0x1FFFFFFull)      // gaddr hi
          | (2u << 30);                                       // type=2 (image)
    v8u g1;
    g1[0] = (1u << 16);                 // wg_mask=0 | data_size=1 (2 bytes)
    g1[1] = (elems & 0xFFFFu) << 16;    // tensor_dim0[15:0]  (bits 63:48)
    g1[2] = ((elems >> 16) & 0xFFFFu)   // tensor_dim0[31:16] (bits 79:64)
          | (1u << 16);                 // tensor_dim1 = 1    (bits 95:80)
    g1[3] = (elems & 0xFFFFu) << 16;    // tile_dim0          (bits 127:112)
    g1[4] = 1u;                         // tile_dim1 = 1, tile_dim2 = 0
    g1[5] = elems;                      // tensor_dim0_stride[31:0]
    g1[6] = 0u;                         // stride hi | tensor_dim1_stride lo
    g1[7] = 0u;
    asm volatile("tensor_load_to_lds %0, %1"
                 :: "s"(g0), "s"(g1)
                 : "memory");
}

// A-matrix fragment (16x32 bf16) from a row-major [rows x ld] buffer.
// lanes 0-15: M=lane, K = kbase+0..15; lanes 16-31: M=lane-16, K=+16.
__device__ __forceinline__ v16bf load_afrag_rm(const __bf16* A, int ld,
                                               int m0, int kbase, int lane) {
    int m   = m0 + (lane & 15);
    int kof = kbase + ((lane >> 4) << 4);
    const __bf16* p = A + (size_t)m * ld + kof;
    v16bf a;
#pragma unroll
    for (int i = 0; i < 8; ++i) a[i] = p[i];
#pragma unroll
    for (int i = 0; i < 8; ++i) a[8 + i] = p[8 + i];
    return a;
}

// B-matrix fragment (32x16 bf16) where column n of B is a contiguous row in
// memory (B[k][n] = Rows[n][k]): one aligned 32-byte load per lane.
// lanes 0-15: N=n0+lane, K=kbase..kbase+15; lanes 16-31: K += 16.
__device__ __forceinline__ v16bf load_bfrag_rows(const __bf16* Rows, int ld,
                                                 int n0, int kbase, int lane) {
    int n = n0 + (lane & 15);
    int k = kbase + ((lane >> 4) << 4);
    return *reinterpret_cast<const v16bf*>(Rows + (size_t)n * ld + k);
}

// ---------------------------------------------------------------------------
// Kernel 1: fp32 -> bf16, producing both row-major xb[b][s][d] and
// transposed xbt[b][d][s] (LDS-tiled 32x32 transpose, coalesced both ways).
// ---------------------------------------------------------------------------
__global__ __launch_bounds__(1024)
void cvt_tr_kernel(const float* __restrict__ x,
                   __bf16* __restrict__ xb, __bf16* __restrict__ xbt) {
    __shared__ __bf16 t[32][33];
    const int b  = blockIdx.z;
    const int s0 = blockIdx.x * 32;
    const int d0 = blockIdx.y * 32;
    const int tx = threadIdx.x, ty = threadIdx.y;

    float v = x[((size_t)b * Sc + s0 + ty) * Dc + d0 + tx];
    __bf16 bv = (__bf16)v;
    xb[((size_t)b * Sc + s0 + ty) * Dc + d0 + tx] = bv;
    t[ty][tx] = bv;
    __syncthreads();
    xbt[((size_t)b * Dc + d0 + ty) * Sc + s0 + tx] = t[tx][ty];
}

// ---------------------------------------------------------------------------
// Kernel 2: dequantize int8-style weight -> bf16:  w[o,d] = (q-128)*scale[o]
// ---------------------------------------------------------------------------
__global__ void dequant_kernel(const int* __restrict__ wq,
                               const float* __restrict__ scale,
                               __bf16* __restrict__ wb, int d) {
    int i = blockIdx.x * blockDim.x + threadIdx.x;
    if (i < d * d) {
        int o = i / d;
        wb[i] = (__bf16)(((float)wq[i] - 128.0f) * scale[o]);
    }
}

// ---------------------------------------------------------------------------
// Kernel 3: fused causal attention for one 16-row query block per workgroup.
// 4 waves / 128 threads. Q tile staged to LDS by the Tensor Data Mover.
// ---------------------------------------------------------------------------
__global__ __launch_bounds__(128)
void attn_kernel(const __bf16* __restrict__ xb,
                 const __bf16* __restrict__ xbt,
                 __bf16* __restrict__ attnb) {
    extern __shared__ char smem[];
    __bf16* Qs    = reinterpret_cast<__bf16*>(smem);                 // [16][Dc]
    __bf16* P     = reinterpret_cast<__bf16*>(smem + 16 * Dc * 2);   // [16][Sc]
    float*  red   = reinterpret_cast<float*>(smem + 16 * Dc * 2 + 16 * Sc * 2);
    float*  stats = red + 16 * 8;

    const int tid  = threadIdx.x;
    const int lane = tid & 31;
    const int wave = tid >> 5;                 // 0..3

    const int blk = blockIdx.x;                // Bc * Sc/16 blocks
    const int b   = blk / (Sc / 16);
    const int q0  = (blk % (Sc / 16)) * 16;

    const __bf16* X  = xb  + (size_t)b * Sc * Dc;   // row-major  [S][D]
    const __bf16* XT = xbt + (size_t)b * Dc * Sc;   // transposed [D][S]
    const int klen   = q0 + 16;
    const int klen32 = (klen + 31) & ~31;
    const int ntiles = klen32 / 16;

    // ---- Phase 0: TDM-stage the contiguous 32KB Q tile into LDS ----------
    if (wave == 0) {
        unsigned int lds_off =
            (unsigned int)(unsigned long long)(lds_raw_t)(void*)Qs;
        tdm_load_to_lds_1d(lds_off,
                           (unsigned long long)(const void*)(X + (size_t)q0 * Dc),
                           16 * Dc);                 // 16384 bf16 elements
        __builtin_amdgcn_s_wait_tensorcnt(0);
    }
    __syncthreads();

    // ---- Phase 1: scores S = (Q K^T) / 32, causal-masked, into LDS -------
    const float inv_sqrt_d = 0.03125f;         // 1/sqrt(1024)
    for (int t = wave; t < ntiles; t += 4) {
        const int c0 = t * 16;
        v8f acc = zero8();
        for (int kb = 0; kb < Dc; kb += 32) {
            v16bf a  = load_afrag_rm(Qs, Dc, 0, kb, lane);      // Q (LDS)
            v16bf bm = load_bfrag_rows(X, Dc, c0, kb, lane);    // K rows
            acc = wmma_bf16(a, bm, acc);
        }
        const int col = lane & 15;
        const int rb  = (lane >> 4) * 8;
#pragma unroll
        for (int r = 0; r < 8; ++r) {
            int   m = rb + r;
            float s = acc[r] * inv_sqrt_d;
            if (c0 + col > q0 + m) s = -1.0e30f;                // causal mask
            P[m * Sc + c0 + col] = (__bf16)s;
        }
    }
    __syncthreads();

    // ---- Phase 2: row softmax over klen32 columns (8 threads per row) ----
    {
        const int row = tid & 15;
        const int sub = tid >> 4;              // 0..7
        float m = -3.0e38f;
        for (int c = sub; c < klen32; c += 8)
            m = fmaxf(m, (float)P[row * Sc + c]);
        red[row * 8 + sub] = m;
        __syncthreads();
        if (sub == 0) {
            float mm = red[row * 8];
#pragma unroll
            for (int i = 1; i < 8; ++i) mm = fmaxf(mm, red[row * 8 + i]);
            stats[row] = mm;
        }
        __syncthreads();
        const float mx = stats[row];
        float s = 0.0f;
        for (int c = sub; c < klen32; c += 8) {
            float e = __expf((float)P[row * Sc + c] - mx);
            P[row * Sc + c] = (__bf16)e;
            s += e;
        }
        __syncthreads();
        red[row * 8 + sub] = s;
        __syncthreads();
        if (sub == 0) {
            float ss = 0.0f;
#pragma unroll
            for (int i = 0; i < 8; ++i) ss += red[row * 8 + i];
            stats[row] = 1.0f / ss;
        }
        __syncthreads();
        const float inv = stats[row];
        for (int c = sub; c < klen32; c += 8)
            P[row * Sc + c] = (__bf16)((float)P[row * Sc + c] * inv);
    }
    __syncthreads();

    // ---- Phase 3: O = P @ V ; each wave owns 16 x 256 of the output ------
    // B-fragment = V columns = contiguous rows of the transposed copy XT.
    const int nbase = wave * 256;
    v8f accs[16];
#pragma unroll
    for (int nt = 0; nt < 16; ++nt) accs[nt] = zero8();

    for (int kb = 0; kb < klen32; kb += 32) {
        v16bf a = load_afrag_rm(P, Sc, 0, kb, lane);            // probs (LDS)
#pragma unroll
        for (int nt = 0; nt < 16; ++nt) {
            v16bf bv = load_bfrag_rows(XT, Sc, nbase + nt * 16, kb, lane);
            accs[nt] = wmma_bf16(a, bv, accs[nt]);
        }
    }

    const int col = lane & 15;
    const int rb  = (lane >> 4) * 8;
#pragma unroll
    for (int nt = 0; nt < 16; ++nt) {
#pragma unroll
        for (int r = 0; r < 8; ++r) {
            attnb[((size_t)b * Sc + q0 + rb + r) * Dc + nbase + nt * 16 + col] =
                (__bf16)accs[nt][r];
        }
    }
}

// ---------------------------------------------------------------------------
// Kernel 4: projection GEMM  out[m, o] = sum_d attn[m, d] * w[o, d]
// A strip (16 x 1024, 32KB) async-staged to LDS, shared by the 4 waves.
// Each wave: 16x64 tile (4 WMMA accumulators).
// ---------------------------------------------------------------------------
__global__ __launch_bounds__(128)
void proj_kernel(const __bf16* __restrict__ attnb,
                 const __bf16* __restrict__ wb,
                 float* __restrict__ out) {
    __shared__ __bf16 As[16 * Dc];                    // 32 KB

    const int tid  = threadIdx.x;
    const int lane = tid & 31;
    const int wave = tid >> 5;

    const int m0 = blockIdx.x * 16;                   // 0 .. B*S-16
    const int n0 = blockIdx.y * 256 + wave * 64;      // 0 .. 1023

    // async-stage contiguous 32KB A strip (GLOBAL_LOAD_ASYNC_TO_LDS_B128)
    {
        const char* gsrc = (const char*)(attnb + (size_t)m0 * Dc);
        char*       ldst = (char*)As;
        for (int i = tid; i < (16 * Dc * 2) / 16; i += 128)
            async_copy_b128(ldst + i * 16, gsrc + i * 16);
        wait_async_zero();
        __syncthreads();
    }

    v8f acc[4];
#pragma unroll
    for (int j = 0; j < 4; ++j) acc[j] = zero8();

    for (int kb = 0; kb < Dc; kb += 32) {
        // prefetch upcoming weight rows (speculative, RT)
        if (kb + 32 < Dc)
            __builtin_prefetch(wb + (size_t)n0 * Dc + kb + 32, 0, 1);
        v16bf a = load_afrag_rm(As, Dc, 0, kb, lane);           // A (LDS)
#pragma unroll
        for (int j = 0; j < 4; ++j) {
            v16bf bw = load_bfrag_rows(wb, Dc, n0 + j * 16, kb, lane);
            acc[j] = wmma_bf16(a, bw, acc[j]);
        }
    }

    const int col = lane & 15;
    const int rb  = (lane >> 4) * 8;
#pragma unroll
    for (int j = 0; j < 4; ++j) {
#pragma unroll
        for (int r = 0; r < 8; ++r) {
            out[(size_t)(m0 + rb + r) * Dc + n0 + j * 16 + col] = acc[j][r];
        }
    }
}

// ---------------------------------------------------------------------------
// Host-side launch
// ---------------------------------------------------------------------------
extern "C" void kernel_launch(void* const* d_in, const int* in_sizes, int n_in,
                              void* d_out, int out_size, void* d_ws, size_t ws_size,
                              hipStream_t stream) {
    const float* x     = (const float*)d_in[0];   // [B,S,D] fp32
    const int*   wq    = (const int*)d_in[1];     // [D,D] int32
    const float* scale = (const float*)d_in[2];   // [D] fp32
    float*       out   = (float*)d_out;           // [B,S,D] fp32

    const int N = Bc * Sc * Dc;                   // 8,388,608

    // Workspace (bf16): xb [B,S,D] | attnb [B,S,D] | wb [D,D] | xbt [B,D,S]
    char* ws = (char*)d_ws;
    __bf16* xb    = (__bf16*)ws;
    __bf16* attnb = (__bf16*)(ws + (size_t)N * 2);
    __bf16* wb    = (__bf16*)(ws + (size_t)N * 4);
    __bf16* xbt   = (__bf16*)(ws + (size_t)N * 4 + (size_t)Dc * Dc * 2);

    // 1) x -> bf16 (row-major + transposed copy)
    {
        dim3 grid(Sc / 32, Dc / 32, Bc);
        dim3 block(32, 32);
        cvt_tr_kernel<<<grid, block, 0, stream>>>(x, xb, xbt);
    }

    // 2) dequantize weight -> bf16
    dequant_kernel<<<(Dc * Dc + 255) / 256, 256, 0, stream>>>(wq, scale, wb, Dc);

    // 3) fused causal attention (one 16-row query block per workgroup)
    {
        const int blocks = Bc * (Sc / 16);                 // 512
        const size_t smem = (size_t)16 * Dc * 2            // Q tile (bf16)
                          + (size_t)16 * Sc * 2            // score strip (bf16)
                          + 16 * 8 * sizeof(float)         // partial reductions
                          + 16 * sizeof(float);            // row stats
        attn_kernel<<<blocks, 128, smem, stream>>>(xb, xbt, attnb);
    }

    // 4) projection GEMM -> fp32 output
    {
        dim3 grid((Bc * Sc) / 16, Dc / 256);               // 512 x 4
        proj_kernel<<<grid, 128, 0, stream>>>(attnb, wb, out);
    }
}
